// Head_86457691668642
// MI455X (gfx1250) — matrix operand order repaired and checked by reading, not compiled, not run on hardware
//
#include <hip/hip_runtime.h>

#define BQ   128      // batches
#define TT   256      // sequence length
#define CE   768      // embedding dim (K of projections)
#define DD   768      // head size (N of projections)
#define MTOT (BQ*TT)  // 32768 rows

typedef __attribute__((ext_vector_type(16))) __bf16 v16bf;
typedef __attribute__((ext_vector_type(8)))  float  v8f;
typedef int v4i __attribute__((vector_size(16)));   // matches async builtin param
typedef unsigned int   u32;
typedef unsigned short u16;

union AFrag { v16bf v; u32 u[8]; };

// ---- fp32 -> packed bf16x2 (hw cvt when declared; else RNE in software) ---
#if __has_builtin(__builtin_amdgcn_cvt_pk_bf16_f32)
__device__ __forceinline__ u32 pack2(float lo, float hi) {
    auto p = __builtin_amdgcn_cvt_pk_bf16_f32(lo, hi);
    return __builtin_bit_cast(u32, p);
}
#else
__device__ __forceinline__ u32 f2bf_bits_(float f) {
    u32 x = __builtin_bit_cast(u32, f);
    return (x + 0x7fffu + ((x >> 16) & 1u)) >> 16;
}
__device__ __forceinline__ u32 pack2(float lo, float hi) {
    return f2bf_bits_(lo) | (f2bf_bits_(hi) << 16);
}
#endif
__device__ __forceinline__ u16 f2bf_u16(float f) { return (u16)(pack2(f, 0.0f) & 0xffffu); }
__device__ __forceinline__ float bf2f(u16 b) {
    return __builtin_bit_cast(float, (u32)b << 16);
}

// ---- async global->LDS staging (CDNA5 GLOBAL_LOAD_ASYNC_TO_LDS_*) --------
#if __has_builtin(__builtin_amdgcn_global_load_async_to_lds_b32) && \
    __has_builtin(__builtin_amdgcn_global_load_async_to_lds_b128)
#define HAVE_ASYNC 1
#define GLB128(p) ((__attribute__((address_space(1))) v4i*)(p))
#define LDS128(p) ((__attribute__((address_space(3))) v4i*)(p))
#if __has_builtin(__builtin_amdgcn_s_wait_asynccnt)
#define WAIT_ASYNC(n) __builtin_amdgcn_s_wait_asynccnt(n)
#else
#define WAIT_ASYNC(n) asm volatile("s_wait_asynccnt %0" :: "i"(n) : "memory")
#endif
#else
#define WAIT_ASYNC(n) ((void)0)
#endif

// ---------------------------------------------------------------------------
// Pre-pass A: x (fp32) -> xbf (bf16), 8 elements/thread, b128 in / b128 out.
// ---------------------------------------------------------------------------
__global__ __launch_bounds__(256)
void cvt_x_kernel(const float* __restrict__ in, u16* __restrict__ outp, long n8)
{
    long i = (long)blockIdx.x * 256 + threadIdx.x;
    if (i >= n8) return;
    const float4* p4 = (const float4*)in + i * 2;
    float4 a = p4[0], b = p4[1];
    uint4 r;
    r.x = pack2(a.x, a.y); r.y = pack2(a.z, a.w);
    r.z = pack2(b.x, b.y); r.w = pack2(b.z, b.w);
    *((uint4*)outp + i) = r;
}

// ---------------------------------------------------------------------------
// Pre-pass B: W (fp32 [k][n]) -> Wt (bf16 [n][k]), one element per thread.
// One-time 768x768x3; coalesced read, scattered u16 write.
// ---------------------------------------------------------------------------
__global__ __launch_bounds__(256)
void cvt_wt_kernel(const float* __restrict__ Wq,
                   const float* __restrict__ Wk,
                   const float* __restrict__ Wv,
                   u16* __restrict__ Wt)
{
    const float* W = (blockIdx.z == 0) ? Wq : (blockIdx.z == 1) ? Wk : Wv;
    u16* o = Wt + (long)blockIdx.z * CE * DD;
    int ty = threadIdx.x >> 4, tx = threadIdx.x & 15;
    int k = blockIdx.y * 16 + ty;     // W row
    int n = blockIdx.x * 16 + tx;     // W col
    o[(long)n * CE + k] = f2bf_u16(W[(long)k * DD + n]);
}

// ---------------------------------------------------------------------------
// Kernel 1: Y = x @ W, all-bf16 datapath with DOUBLE-BUFFERED async staging:
// issue async batch for tile k+1 into the alternate LDS buffer, then
// s_wait_asynccnt 3 (3 copies/thread/batch, in-order completion => previous
// batch landed), barrier, and compute tile k while k+1 streams in.
// z==2 (V) output is stored transposed per batch: Vt[b][d][key].
// ---------------------------------------------------------------------------
#define SAW_H 40   // u16 row stride of x tile  (80 B: keeps b128 align)
#define SBW_H 40   // u16 row stride of Wt tile

__global__ __launch_bounds__(256)
void qkv_gemm_kernel(const u16* __restrict__ xbf,
                     const u16* __restrict__ Wt,   // 3 transposed mats [n][k]
                     u16* __restrict__ Qb,
                     u16* __restrict__ Kb,
                     u16* __restrict__ Vt)
{
    const u16* W = Wt + (long)blockIdx.z * CE * DD;

    __shared__ u16 sA[2][128 * SAW_H];   // x tile  [m][k] bf16, double buffer
    __shared__ u16 sB[2][ 64 * SBW_H];   // W tile  [n][k] bf16, double buffer

    const int t    = threadIdx.x;
    const int lane = t & 31;
    const int wave = t >> 5;
    const int wm   = wave >> 1;        // 0..3   (M sub-block)
    const int wn   = wave & 1;         // 0..1   (N sub-block)
    const int h    = lane >> 4;        // lane half
    const int l16  = lane & 15;

    const int mBlock = blockIdx.y * 128;
    const int nBlock = blockIdx.x * 64;

    // issue one staging batch (3 x b128 per thread) into buffer `buf`
    auto stage = [&](int buf, int k0) {
#ifdef HAVE_ASYNC
        #pragma unroll
        for (int r = 0; r < 2; ++r) {
            int idx = r * 256 + t;                  // 0..511 (8 halves each)
            int mm = idx >> 2, c8 = (idx & 3) << 3;
            __builtin_amdgcn_global_load_async_to_lds_b128(
                GLB128(&xbf[(long)(mBlock + mm) * CE + k0 + c8]),
                LDS128(&sA[buf][mm * SAW_H + c8]), 0, 0);
        }
        {
            int nn = t >> 2, c8 = (t & 3) << 3;
            __builtin_amdgcn_global_load_async_to_lds_b128(
                GLB128(&W[(long)(nBlock + nn) * CE + k0 + c8]),
                LDS128(&sB[buf][nn * SBW_H + c8]), 0, 0);
        }
#else
        #pragma unroll
        for (int r = 0; r < 8; ++r) {               // 2048 u32 of x tile
            int idx = r * 256 + t;
            int mm = idx >> 4, cp = (idx & 15) << 1;
            *(u32*)&sA[buf][mm * SAW_H + cp] =
                *(const u32*)&xbf[(long)(mBlock + mm) * CE + k0 + cp];
        }
        #pragma unroll
        for (int r = 0; r < 4; ++r) {               // 1024 u32 of Wt tile
            int idx = r * 256 + t;
            int nn = idx >> 4, cp = (idx & 15) << 1;
            *(u32*)&sB[buf][nn * SBW_H + cp] =
                *(const u32*)&W[(long)(nBlock + nn) * CE + k0 + cp];
        }
#endif
    };

    v8f c00 = {}, c01 = {}, c10 = {}, c11 = {};

    stage(0, 0);                                   // prologue batch
    int ib = 0;
    for (int k0 = 0; k0 < CE; k0 += 32, ib ^= 1) {
        if (k0 + 32 < CE) {
            stage(ib ^ 1, k0 + 32);                // overlap next tile's copy
            WAIT_ASYNC(3);                         // previous batch complete
        } else {
            WAIT_ASYNC(0);
        }
        // warm L2 two tiles ahead (global_prefetch_b8)
        if (k0 + 64 < CE)
            __builtin_prefetch(&xbf[(long)(mBlock + (t >> 1)) * CE + k0 + 64], 0, 1);
        __syncthreads();                           // all waves' buffers ready

        AFrag a0, a1, b0, b1;
        #pragma unroll
        for (int i = 0; i < 8; ++i) {
            int ka = ((i >> 2) << 4) + (h << 3) + ((i & 3) << 1);  // A K-pair
            int kb = ((i >> 2) << 4) + (h << 3) + ((i & 3) << 1);  // B K-pair
            a0.u[i] = *(const u32*)&sA[ib][(wm * 32 +      l16) * SAW_H + ka];
            a1.u[i] = *(const u32*)&sA[ib][(wm * 32 + 16 + l16) * SAW_H + ka];
            b0.u[i] = *(const u32*)&sB[ib][(wn * 32 +      l16) * SBW_H + kb];
            b1.u[i] = *(const u32*)&sB[ib][(wn * 32 + 16 + l16) * SBW_H + kb];
        }
        c00 = __builtin_amdgcn_wmma_f32_16x16x32_bf16(false, a0.v, false, b0.v, (short)0, c00, false, false);
        c01 = __builtin_amdgcn_wmma_f32_16x16x32_bf16(false, a0.v, false, b1.v, (short)0, c01, false, false);
        c10 = __builtin_amdgcn_wmma_f32_16x16x32_bf16(false, a1.v, false, b0.v, (short)0, c10, false, false);
        c11 = __builtin_amdgcn_wmma_f32_16x16x32_bf16(false, a1.v, false, b1.v, (short)0, c11, false, false);
        __syncthreads();                           // reads done before rewrite
    }

    // write back as bf16 (C/D layout: lane -> column, VGPR r + 8h -> row)
    const v8f cs[4] = { c00, c01, c10, c11 };
    u16* Y = (blockIdx.z == 0) ? Qb : (blockIdx.z == 1) ? Kb : Vt;
    #pragma unroll
    for (int mi = 0; mi < 2; ++mi) {
        #pragma unroll
        for (int ni = 0; ni < 2; ++ni) {
            const v8f c = cs[mi * 2 + ni];
            int n = nBlock + wn * 32 + ni * 16 + l16;
            #pragma unroll
            for (int r = 0; r < 8; ++r) {
                int m = mBlock + wm * 32 + mi * 16 + r + h * 8;
                if (blockIdx.z == 2) {
                    int bb = m >> 8, tp = m & 255;                 // TT == 256
                    Y[((long)bb * DD + n) * TT + tp] = f2bf_u16(c[r]);
                } else {
                    Y[(long)m * DD + n] = f2bf_u16(c[r]);
                }
            }
        }
    }
}

// ---------------------------------------------------------------------------
// Kernel 2: in-place RoPE on Q (z=0) and K (z=1); one (2j,2j+1) pair/thread.
// ---------------------------------------------------------------------------
__global__ __launch_bounds__(256)
void rope_kernel(u16* __restrict__ Qb, u16* __restrict__ Kb)
{
    u16* Yp = (blockIdx.z == 0) ? Qb : Kb;
    long gid = (long)blockIdx.x * 256 + threadIdx.x;      // pair index
    const long npairs = (long)MTOT * (DD / 2);
    if (gid >= npairs) return;
    int  j    = (int)(gid % (DD / 2));
    long row  = gid / (DD / 2);
    int  tpos = (int)(row % TT);
    float theta = __expf((float)j * (-2.0f * 9.210340371976184f / (float)DD));
    float ang   = (float)tpos * theta;
    float s, c;
    __sincosf(ang, &s, &c);
    u32* p = (u32*)(Yp + row * DD + 2 * j);
    u32 w = *p;
    float re = bf2f((u16)(w & 0xffffu)), im = bf2f((u16)(w >> 16));
    *p = pack2(re * c - im * s, re * s + im * c);
}

// ---------------------------------------------------------------------------
// Kernel 3: attention, one batch per workgroup. Full 256x256 score matrix in
// LDS (f32, stride 260 -> 266,240 B; fits CDNA5's 320 KB WGP LDS). After the
// softmax, P is re-packed to bf16 pairs IN PLACE so phase-3 A-fragments are
// plain ds_load_b32 with zero converts in the inner loop.
// ---------------------------------------------------------------------------
#define SSTR 260

__global__ __launch_bounds__(256)
void attn_kernel(const u16* __restrict__ Qb,
                 const u16* __restrict__ Kb,
                 const u16* __restrict__ Vt,   // [b][d][key]
                 float* __restrict__ Out)
{
    extern __shared__ float S[];                 // [256][SSTR]
    const int b    = blockIdx.x;
    const int t    = threadIdx.x;
    const int lane = t & 31;
    const int wave = t >> 5;
    const int h    = lane >> 4;
    const int l16  = lane & 15;

    const u16* Qbase = Qb + (long)b * TT * DD;
    const u16* Kbase = Kb + (long)b * TT * DD;
    const u16* Vbase = Vt + (long)b * DD * TT;
    const float scale = 0.03608439182435161f;    // 1/sqrt(768)

    // ---- Phase 1: S = scale * Q K^T, causal (lower-triangular) tiles ----
    for (int tile = wave; tile < 256; tile += 8) {
        int qi = tile >> 4, ki = tile & 15;
        if (ki > qi) continue;                   // fully masked tile
        v8f c = {};
        const u16* qrow = Qbase + (long)(qi * 16 + l16) * DD;  // A: lane = row
        const u16* krow = Kbase + (long)(ki * 16 + l16) * DD;  // B: lane = key col
        for (int d0 = 0; d0 < DD; d0 += 32) {
            AFrag a, bf;
            #pragma unroll
            for (int i = 0; i < 8; ++i) {
                int ka = d0 + ((i >> 2) << 4) + (h << 3) + ((i & 3) << 1);
                int kb = d0 + ((i >> 2) << 4) + (h << 3) + ((i & 3) << 1);
                a.u[i]  = *(const u32*)&qrow[ka];
                bf.u[i] = *(const u32*)&krow[kb];   // K row-major == B of K^T
            }
            c = __builtin_amdgcn_wmma_f32_16x16x32_bf16(false, a.v, false, bf.v, (short)0, c, false, false);
        }
        #pragma unroll
        for (int r = 0; r < 8; ++r) {
            int q = qi * 16 + r + h * 8;
            S[q * SSTR + ki * 16 + l16] = c[r] * scale;
        }
    }
    __syncthreads();

    // ---- Phase 2: causal softmax + in-place bf16 pair pack, 1 row/thread ----
    {
        float* row = S + t * SSTR;
        const int kmax = t;                      // keys 0..t valid
        float m = -3.4e38f;
        for (int k = 0; k <= kmax; ++k) m = fmaxf(m, row[k]);
        float sum = 0.f;
        for (int k = 0; k <= kmax; ++k) { float e = __expf(row[k] - m); row[k] = e; sum += e; }
        float inv = 1.0f / sum;
        for (int k = 0; k <= kmax; ++k) row[k] *= inv;
        for (int k = kmax + 1; k < TT; ++k) row[k] = 0.0f;
        // pack: float pair (k,k+1) -> bf16x2 at uint index k/2 (k/2 <= k: safe)
        u32* prow = (u32*)row;
        for (int k = 0; k < TT; k += 2) prow[k >> 1] = pack2(row[k], row[k + 1]);
    }
    __syncthreads();

    // ---- Phase 3: O = P @ V ; 16x16 output tiles over [256 x 768] ----
    const u32* Sp = (const u32*)S;               // packed P rows, uint stride SSTR
    for (int tile = wave; tile < 16 * (DD / 16); tile += 8) {
        int qi = tile / (DD / 16);
        int ni = tile % (DD / 16);
        v8f c = {};
        const u32* prow = Sp + (qi * 16 + l16) * SSTR;          // A: lane = row
        const u16* vcol = Vbase + (long)(ni * 16 + l16) * TT;   // B: lane = d col
        for (int kk0 = 0; kk0 < TT; kk0 += 32) {
            AFrag a, bf;
            #pragma unroll
            for (int i = 0; i < 8; ++i) {
                int ka = kk0 + ((i >> 2) << 4) + (h << 3) + ((i & 3) << 1);
                int kb = kk0 + ((i >> 2) << 4) + (h << 3) + ((i & 3) << 1);
                a.u[i]  = prow[ka >> 1];                 // ds_load_b32
                bf.u[i] = *(const u32*)&vcol[kb];        // global b32 (V transposed)
            }
            c = __builtin_amdgcn_wmma_f32_16x16x32_bf16(false, a.v, false, bf.v, (short)0, c, false, false);
        }
        #pragma unroll
        for (int r = 0; r < 8; ++r) {
            int q = qi * 16 + r + h * 8;
            Out[((long)b * TT + q) * DD + ni * 16 + l16] = c[r];
        }
    }
}

// ---------------------------------------------------------------------------
extern "C" void kernel_launch(void* const* d_in, const int* in_sizes, int n_in,
                              void* d_out, int out_size, void* d_ws, size_t ws_size,
                              hipStream_t stream)
{
    (void)in_sizes; (void)n_in; (void)out_size; (void)ws_size;
    const float* x  = (const float*)d_in[0];
    const float* Wq = (const float*)d_in[1];
    const float* Wk = (const float*)d_in[2];
    const float* Wv = (const float*)d_in[3];
    float* out = (float*)d_out;

    u16* Qb  = (u16*)d_ws;                         // bf16, 48 MB each
    u16* Kb  = Qb  + (size_t)MTOT * DD;
    u16* Vt  = Kb  + (size_t)MTOT * DD;            // transposed [b][d][key]
    u16* xbf = Vt  + (size_t)MTOT * DD;            // bf16 copy of x, 48 MB
    u16* Wt  = xbf + (size_t)MTOT * CE;            // 3 transposed weights, 3.5 MB

    // 0) one-time bf16 conversion pre-passes
    long n8 = (long)MTOT * CE / 8;
    cvt_x_kernel<<<dim3((unsigned)((n8 + 255) / 256)), 256, 0, stream>>>(x, xbf, n8);
    cvt_wt_kernel<<<dim3(DD / 16, CE / 16, 3), 256, 0, stream>>>(Wq, Wk, Wv, Wt);

    // 1) QKV projections (WMMA bf16, double-buffered async LDS staging)
    qkv_gemm_kernel<<<dim3(DD / 64, MTOT / 128, 3), 256, 0, stream>>>(
        xbf, Wt, Qb, Kb, Vt);

    // 2) RoPE on Q and K (in place)
    long npairs = (long)MTOT * (DD / 2);
    rope_kernel<<<dim3((unsigned)((npairs + 255) / 256), 1, 2), 256, 0, stream>>>(Qb, Kb);

    // 3) Attention, one batch per workgroup; 266 KB dynamic LDS for S
    size_t smem = (size_t)TT * SSTR * sizeof(float);
    (void)hipFuncSetAttribute(reinterpret_cast<const void*>(attn_kernel),
                              hipFuncAttributeMaxDynamicSharedMemorySize, (int)smem);
    attn_kernel<<<dim3(BQ), 256, smem, stream>>>(Qb, Kb, Vt, out);
}
